// GlobalLocalEncoder_15710990369676
// MI455X (gfx1250) — compile-verified, matrix operand order checked by reading, not run on hardware
//
#include <hip/hip_runtime.h>

#define E_EDGES 100000
#define NNODES  50000
#define HIDD    128
#define SEQL    4

typedef __attribute__((ext_vector_type(16))) _Float16 v16h;
typedef __attribute__((ext_vector_type(8)))  _Float16 v8h;
typedef __attribute__((ext_vector_type(8)))  float    v8f;

static __device__ __forceinline__ v8f zero8() {
  v8f z;
#pragma unroll
  for (int i = 0; i < 8; ++i) z[i] = 0.0f;
  return z;
}

static __device__ __forceinline__ v16h make_frag(v8h lo, v8h hi) {
  v16h r;
#pragma unroll
  for (int i = 0; i < 8; ++i) { r[i] = lo[i]; r[8 + i] = hi[i]; }
  return r;
}

// convert 8 consecutive f32 -> 8 f16 (two aligned float4 loads)
static __device__ __forceinline__ v8h cvt8(const float* p) {
  float4 a = *(const float4*)p;
  float4 b = *(const float4*)(p + 4);
  v8h r;
  r[0] = (_Float16)a.x; r[1] = (_Float16)a.y; r[2] = (_Float16)a.z; r[3] = (_Float16)a.w;
  r[4] = (_Float16)b.x; r[5] = (_Float16)b.y; r[6] = (_Float16)b.z; r[7] = (_Float16)b.w;
  return r;
}

// 16x32 f16 A/B fragment from a row-major f32 row: p = &row[m][k0]; lane holds
// K-runs [hs*8, hs*8+8) and [16+hs*8, 16+hs*8+8)  (per CDNA5 ISA layout)
static __device__ __forceinline__ v16h afrag_f32(const float* p, int hs) {
  return make_frag(cvt8(p + hs * 8), cvt8(p + 16 + hs * 8));
}

// same fragment layout, source already f16 row-major: p = &row[m-or-n][k0]
static __device__ __forceinline__ v16h frag_f16(const _Float16* p, int hs) {
  return make_frag(*(const v8h*)(p + hs * 8), *(const v8h*)(p + 16 + hs * 8));
}

static __device__ __forceinline__ v8f wmma32(v16h a, v16h b, v8f c) {
  return __builtin_amdgcn_wmma_f32_16x16x32_f16(false, a, false, b, (short)0, c,
                                                false, false);
}

// ---------------------------------------------------------------------------
// prep kernels
// ---------------------------------------------------------------------------
__global__ void zero_pool_kernel(float* acc) { acc[threadIdx.x] = 0.0f; }

// concat f32 weights -> f16 workspace: wq_l|wk_l|wv_l|wo_l|wq_g|wo_g
__global__ void wcvt_kernel(const float* wq_l, const float* wk_l,
                            const float* wv_l, const float* wo_l,
                            const float* wq_g, const float* wo_g,
                            _Float16* w16) {
  int i = blockIdx.x * blockDim.x + threadIdx.x;
  if (i >= 131072) return;
  const float* src; int off;
  if (i < 16384)       { src = wq_l; off = i; }
  else if (i < 49152)  { src = wk_l; off = i - 16384; }
  else if (i < 81920)  { src = wv_l; off = i - 49152; }
  else if (i < 98304)  { src = wo_l; off = i - 81920; }
  else if (i < 114688) { src = wq_g; off = i - 98304; }
  else                 { src = wo_g; off = i - 114688; }
  w16[i] = (_Float16)src[off];
}

// column sums of global_feats viewed as (NNODES, 512)
__global__ void pool_kernel(const float* __restrict__ g, float* acc) {
  int col = threadIdx.x;  // 512 threads
  float s = 0.0f;
  for (int n = blockIdx.x; n < NNODES; n += gridDim.x)
    s += g[(size_t)n * 512 + col];
  atomicAdd(&acc[col], s);
}

// pooled mean, k_g, v_g; write pooled output
__global__ void finalize_kernel(const float* __restrict__ acc,
                                const float* __restrict__ wk_g,
                                const float* __restrict__ wv_g,
                                const float* __restrict__ bk_g,
                                const float* __restrict__ bv_g,
                                float* kg, float* vg, float* out) {
  __shared__ float pooled[512];
  int t = threadIdx.x;  // 128 threads
  for (int i = t; i < 512; i += 128) pooled[i] = acc[i] * (1.0f / (float)NNODES);
  __syncthreads();
  const size_t pool_out = (size_t)E_EDGES * 384;
  for (int i = t; i < 512; i += 128) out[pool_out + i] = pooled[i];
#pragma unroll
  for (int s = 0; s < SEQL; ++s) {
    float dk = bk_g[t], dv = bv_g[t];
    for (int h = 0; h < HIDD; ++h) {
      float p = pooled[s * HIDD + h];
      dk += p * wk_g[t * HIDD + h];
      dv += p * wv_g[t * HIDD + h];
    }
    kg[s * HIDD + t] = dk;
    vg[s * HIDD + t] = dv;
  }
}

// ---------------------------------------------------------------------------
// main: 4 waves per 16-edge tile.
//   wave w: seq-stripe w of the k_l/v_l GEMMs (128 WMMAs), and N-tiles
//           {2w, 2w+1} of the q_l / q_g / out_l / out_g projections.
// All GEMMs via v_wmma_f32_16x16x32_f16 (f16 in, f32 accumulate).
// ---------------------------------------------------------------------------
__global__ __launch_bounds__(128)
void encoder_kernel(const float* __restrict__ edge_attr,
                    const float* __restrict__ gfeat,
                    const long long* __restrict__ eidx,
                    const float* __restrict__ bq_l, const float* __restrict__ bk_l,
                    const float* __restrict__ bv_l, const float* __restrict__ bo_l,
                    const float* __restrict__ bq_g, const float* __restrict__ bo_g,
                    const _Float16* __restrict__ w16,
                    const float* __restrict__ kg, const float* __restrict__ vg,
                    float* __restrict__ out) {
  __shared__ __align__(16) char smem[46080];
  float*    q_lds = (float*)smem;                        // 16x128 f32
  _Float16* k16   = (_Float16*)(smem + 8192);            // 64x128 f16 [(s*16+e)*128+h]
  _Float16* v16   = (_Float16*)(smem + 8192 + 16384);    // 64x128 f16
  _Float16* ao16  = (_Float16*)(smem + 8192 + 32768);    // 16x128 f16
  float*    sc    = (float*)(smem + 8192 + 32768 + 4096);// 64
  float*    aw    = sc + 64;                             // 64
  int*      sidx  = (int*)(aw + 64);                     // 16
  int*      didx  = sidx + 16;                           // 16

  const int tid  = threadIdx.x;
  const int lane = tid & 31;
  const int w    = tid >> 5;        // wave id 0..3  == seq stripe
  const int nlo  = lane & 15;
  const int hs   = lane >> 4;
  const int e0   = blockIdx.x * 16;
  const float scale = 0.08838834764831845f;  // 1/sqrt(128)

  const _Float16* wq_l16 = w16;
  const _Float16* wk_l16 = w16 + 16384;
  const _Float16* wv_l16 = w16 + 49152;
  const _Float16* wo_l16 = w16 + 81920;
  const _Float16* wq_g16 = w16 + 98304;
  const _Float16* wo_g16 = w16 + 114688;

  if (tid < 16) {
    sidx[tid] = (int)eidx[e0 + tid];
    didx[tid] = (int)eidx[E_EDGES + e0 + tid];
  }
  __syncthreads();

  // warm WGP$/L2 for this wave's gather rows (global_prefetch_b8)
  if (lane < 16) {
    __builtin_prefetch(gfeat + ((size_t)sidx[lane] * SEQL + w) * HIDD, 0, 3);
    __builtin_prefetch(gfeat + ((size_t)didx[lane] * SEQL + w) * HIDD, 0, 3);
  }

  // ---- P1: k_l / v_l stripe s=w  (16x256x128 GEMM pair per wave) ----
  {
    const int s = w;
    v8f kacc[8], vacc[8];
#pragma unroll
    for (int t = 0; t < 8; ++t) { kacc[t] = zero8(); vacc[t] = zero8(); }
    for (int kt = 0; kt < 8; ++kt) {
      const int k0 = kt * 32;
      const int node = (k0 < 128) ? sidx[nlo] : didx[nlo];
      const float* rowp = gfeat + ((size_t)node * SEQL + s) * HIDD + (k0 & 127);
      v16h a = afrag_f32(rowp, hs);
#pragma unroll
      for (int nt = 0; nt < 8; ++nt) {
        v16h bk = frag_f16(wk_l16 + (size_t)(nt * 16 + nlo) * 256 + k0, hs);
        kacc[nt] = wmma32(a, bk, kacc[nt]);
        v16h bv = frag_f16(wv_l16 + (size_t)(nt * 16 + nlo) * 256 + k0, hs);
        vacc[nt] = wmma32(a, bv, vacc[nt]);
      }
    }
#pragma unroll
    for (int nt = 0; nt < 8; ++nt) {
      const int n = nt * 16 + nlo;
      const float bk = bk_l[n], bv = bv_l[n];
#pragma unroll
      for (int r = 0; r < 8; ++r) {
        const int m = r + hs * 8;
        k16[(s * 16 + m) * HIDD + n] = (_Float16)(kacc[nt][r] + bk);
        v16[(s * 16 + m) * HIDD + n] = (_Float16)(vacc[nt][r] + bv);
      }
    }
  }

  // ---- P2: q_l = edge_attr @ wq_l^T + bq_l (each wave: 2 N-tiles) ----
  {
    v8f qacc[2];
    qacc[0] = zero8(); qacc[1] = zero8();
    for (int kt = 0; kt < 4; ++kt) {
      const int k0 = kt * 32;
      v16h a = afrag_f32(edge_attr + (size_t)(e0 + nlo) * HIDD + k0, hs);
#pragma unroll
      for (int j = 0; j < 2; ++j) {
        const int nt = w * 2 + j;
        v16h b = frag_f16(wq_l16 + (size_t)(nt * 16 + nlo) * 128 + k0, hs);
        qacc[j] = wmma32(a, b, qacc[j]);
      }
    }
#pragma unroll
    for (int j = 0; j < 2; ++j) {
      const int n = (w * 2 + j) * 16 + nlo;
      const float bq = bq_l[n];
#pragma unroll
      for (int r = 0; r < 8; ++r) q_lds[(r + hs * 8) * HIDD + n] = qacc[j][r] + bq;
    }
  }
  __syncthreads();

  // ---- P3: local attention (VALU, 128 threads) ----
  if (tid < 64) {
    const int e = tid >> 2, s = tid & 3;
    float d = 0.0f;
    for (int h = 0; h < HIDD; ++h)
      d += q_lds[e * HIDD + h] * (float)k16[(s * 16 + e) * HIDD + h];
    sc[tid] = d * scale;
  }
  __syncthreads();
  if (tid < 16) {
    float m = sc[tid * 4];
#pragma unroll
    for (int s = 1; s < 4; ++s) m = fmaxf(m, sc[tid * 4 + s]);
    float ex[4], sum = 0.0f;
#pragma unroll
    for (int s = 0; s < 4; ++s) { ex[s] = __expf(sc[tid * 4 + s] - m); sum += ex[s]; }
    const float inv = 1.0f / sum;
#pragma unroll
    for (int s = 0; s < 4; ++s) aw[tid * 4 + s] = ex[s] * inv;
  }
  __syncthreads();
  for (int i = tid; i < 2048; i += 128) {
    const int e = i >> 7, h = i & 127;
    float acc = 0.0f;
#pragma unroll
    for (int s = 0; s < 4; ++s)
      acc += aw[e * 4 + s] * (float)v16[(s * 16 + e) * HIDD + h];
    ao16[e * HIDD + h] = (_Float16)acc;
  }
  __syncthreads();

  // ---- P4: local_out = ao @ wo_l^T + bo_l ; plus edge_attr passthrough ----
  {
    v8f oacc[2];
    oacc[0] = zero8(); oacc[1] = zero8();
    for (int kt = 0; kt < 4; ++kt) {
      const int k0 = kt * 32;
      v16h a = frag_f16(ao16 + nlo * HIDD + k0, hs);
#pragma unroll
      for (int j = 0; j < 2; ++j) {
        const int nt = w * 2 + j;
        v16h b = frag_f16(wo_l16 + (size_t)(nt * 16 + nlo) * 128 + k0, hs);
        oacc[j] = wmma32(a, b, oacc[j]);
      }
    }
#pragma unroll
    for (int j = 0; j < 2; ++j) {
      const int n = (w * 2 + j) * 16 + nlo;
      const float bo = bo_l[n];
#pragma unroll
      for (int r = 0; r < 8; ++r) {
        const int m = r + hs * 8;
        out[(size_t)(e0 + m) * 384 + 128 + n] = oacc[j][r] + bo;
      }
    }
  }
  for (int i = tid; i < 2048; i += 128) {
    const int e = i >> 7, h = i & 127;
    out[(size_t)(e0 + e) * 384 + h] = edge_attr[(size_t)(e0 + e) * HIDD + h];
  }
  __syncthreads();   // ao16 / q_lds about to be overwritten

  // ---- P5: q_g = edge_attr @ wq_g^T + bq_g (reuse q_lds) ----
  {
    v8f qacc[2];
    qacc[0] = zero8(); qacc[1] = zero8();
    for (int kt = 0; kt < 4; ++kt) {
      const int k0 = kt * 32;
      v16h a = afrag_f32(edge_attr + (size_t)(e0 + nlo) * HIDD + k0, hs);
#pragma unroll
      for (int j = 0; j < 2; ++j) {
        const int nt = w * 2 + j;
        v16h b = frag_f16(wq_g16 + (size_t)(nt * 16 + nlo) * 128 + k0, hs);
        qacc[j] = wmma32(a, b, qacc[j]);
      }
    }
#pragma unroll
    for (int j = 0; j < 2; ++j) {
      const int n = (w * 2 + j) * 16 + nlo;
      const float bq = bq_g[n];
#pragma unroll
      for (int r = 0; r < 8; ++r) q_lds[(r + hs * 8) * HIDD + n] = qacc[j][r] + bq;
    }
  }
  __syncthreads();

  // ---- P6: global attention (VALU, k_g/v_g precomputed in ws) ----
  if (tid < 64) {
    const int e = tid >> 2, s = tid & 3;
    float d = 0.0f;
    for (int h = 0; h < HIDD; ++h) d += q_lds[e * HIDD + h] * kg[s * HIDD + h];
    sc[tid] = d * scale;
  }
  __syncthreads();
  if (tid < 16) {
    float m = sc[tid * 4];
#pragma unroll
    for (int s = 1; s < 4; ++s) m = fmaxf(m, sc[tid * 4 + s]);
    float ex[4], sum = 0.0f;
#pragma unroll
    for (int s = 0; s < 4; ++s) { ex[s] = __expf(sc[tid * 4 + s] - m); sum += ex[s]; }
    const float inv = 1.0f / sum;
    const size_t awbase = (size_t)E_EDGES * 384 + 512;
#pragma unroll
    for (int s = 0; s < 4; ++s) {
      const float wgt = ex[s] * inv;
      aw[tid * 4 + s] = wgt;
      out[awbase + (size_t)(e0 + tid) * 4 + s] = wgt;  // global_attn_weights
    }
  }
  __syncthreads();
  for (int i = tid; i < 2048; i += 128) {
    const int e = i >> 7, h = i & 127;
    float acc = 0.0f;
#pragma unroll
    for (int s = 0; s < 4; ++s) acc += aw[e * 4 + s] * vg[s * HIDD + h];
    ao16[e * HIDD + h] = (_Float16)acc;
  }
  __syncthreads();

  // ---- P7: global_out = go @ wo_g^T + bo_g ----
  {
    v8f oacc[2];
    oacc[0] = zero8(); oacc[1] = zero8();
    for (int kt = 0; kt < 4; ++kt) {
      const int k0 = kt * 32;
      v16h a = frag_f16(ao16 + nlo * HIDD + k0, hs);
#pragma unroll
      for (int j = 0; j < 2; ++j) {
        const int nt = w * 2 + j;
        v16h b = frag_f16(wo_g16 + (size_t)(nt * 16 + nlo) * 128 + k0, hs);
        oacc[j] = wmma32(a, b, oacc[j]);
      }
    }
#pragma unroll
    for (int j = 0; j < 2; ++j) {
      const int n = (w * 2 + j) * 16 + nlo;
      const float bo = bo_g[n];
#pragma unroll
      for (int r = 0; r < 8; ++r) {
        const int m = r + hs * 8;
        out[(size_t)(e0 + m) * 384 + 256 + n] = oacc[j][r] + bo;
      }
    }
  }
}

// ---------------------------------------------------------------------------
extern "C" void kernel_launch(void* const* d_in, const int* in_sizes, int n_in,
                              void* d_out, int out_size, void* d_ws, size_t ws_size,
                              hipStream_t stream) {
  const float*     edge_attr = (const float*)d_in[0];
  const float*     gfeat     = (const float*)d_in[1];
  const long long* eidx      = (const long long*)d_in[2];
  const float*     wq_l = (const float*)d_in[3];
  const float*     wk_l = (const float*)d_in[4];
  const float*     wv_l = (const float*)d_in[5];
  const float*     bq_l = (const float*)d_in[6];
  const float*     bk_l = (const float*)d_in[7];
  const float*     bv_l = (const float*)d_in[8];
  const float*     wo_l = (const float*)d_in[9];
  const float*     bo_l = (const float*)d_in[10];
  const float*     wq_g = (const float*)d_in[11];
  const float*     wk_g = (const float*)d_in[12];
  const float*     wv_g = (const float*)d_in[13];
  const float*     bq_g = (const float*)d_in[14];
  const float*     bk_g = (const float*)d_in[15];
  const float*     bv_g = (const float*)d_in[16];
  const float*     wo_g = (const float*)d_in[17];
  const float*     bo_g = (const float*)d_in[18];
  float* out = (float*)d_out;

  // workspace carve: pool_acc(512 f32) | k_g(512 f32) | v_g(512 f32) | w16(131072 f16)
  float*    pool_acc = (float*)d_ws;
  float*    kg       = pool_acc + 512;
  float*    vg       = kg + 512;
  _Float16* w16      = (_Float16*)(vg + 512);

  zero_pool_kernel<<<1, 512, 0, stream>>>(pool_acc);
  wcvt_kernel<<<512, 256, 0, stream>>>(wq_l, wk_l, wv_l, wo_l, wq_g, wo_g, w16);
  pool_kernel<<<128, 512, 0, stream>>>(gfeat, pool_acc);
  finalize_kernel<<<1, 128, 0, stream>>>(pool_acc, wk_g, wv_g, bk_g, bv_g, kg, vg, out);
  encoder_kernel<<<E_EDGES / 16, 128, 0, stream>>>(
      edge_attr, gfeat, eidx, bq_l, bk_l, bv_l, bo_l, bq_g, bo_g,
      w16, kg, vg, out);
}
